// GravityNet_81604378624570
// MI455X (gfx1250) — matrix operand order repaired and checked by reading, not compiled, not run on hardware
//
#include <hip/hip_runtime.h>

// ---------------------------------------------------------------------------
// GravityNet pair-conv on gfx1250: radial-basis MLP as f16 WMMA batched GEMM.
//   layer0: [128 pairs, 32(K pad of 30)] x [32, 112(N pad of 100)]   7 WMMA
//   layer1: [128, 128(K pad)] x [128, 112(N pad)]                   28 WMMA
//   layer2: dot(112) with w2 (padded), then Y * R * feat accumulation.
// One workgroup (256 thr = 8 waves, wave32) per (z, a); 8 chunks of 128 b's.
// sAct columns 112..127 are zero-filled once (K-padding for layer1) so no
// uninitialized-LDS NaNs can propagate through WMMA.
// ---------------------------------------------------------------------------

typedef _Float16 v8h  __attribute__((ext_vector_type(8)));
typedef _Float16 v16h __attribute__((ext_vector_type(16)));
typedef float    v8f  __attribute__((ext_vector_type(8)));

#define NT   7     // N tiles: 112 hidden columns (pad of 100)
#define SB   40    // LDS stride (halfs) for sBasis / sW0T  (80B, 16B aligned)
#define SA   136   // LDS stride (halfs) for sAct           (272B, 16B aligned)
#define RS30  0.18257418583f   // 1/sqrt(30)
#define RS100 0.1f             // 1/sqrt(100)
#define SQRT3 1.7320508075f
// SP_FACTOR for beta=5 rescaled softplus: 1/sqrt(E[(softplus(5x)/5)^2]), x~N(0,1)
#define SPK   (1.4030f / 5.0f) // SP_FACTOR / BETA

__device__ __forceinline__ float actf(float x) {
    float y = 5.0f * x;                       // BETA * x
    // stable softplus: max(y,0) + log1p(exp(-|y|))
    float sp = fmaxf(y, 0.0f) + __logf(1.0f + __expf(-fabsf(y)));
    return sp * SPK;
}

// prep: W1 [100,100] f32 -> W1T [128,128] f16 (transposed, scaled, zero-padded)
__global__ void grav_prep_w1(const float* __restrict__ W1,
                             _Float16* __restrict__ W1T) {
    int i = blockIdx.x * blockDim.x + threadIdx.x;   // 0 .. 16383
    int n = i >> 7, k = i & 127;
    float v = (n < 100 && k < 100) ? W1[k * 100 + n] * RS100 : 0.0f;
    W1T[n * 128 + k] = (_Float16)v;
}

union AF { v16h v; v8h h[2]; };

__global__ __launch_bounds__(256)
void grav_main(const float* __restrict__ features,
               const float* __restrict__ geometry,
               const float* __restrict__ W0,
               const float* __restrict__ W2,
               const _Float16* __restrict__ W1T,
               float* __restrict__ out) {
    __shared__ __attribute__((aligned(16))) _Float16 sW0T[112 * SB];
    __shared__ __attribute__((aligned(16))) _Float16 sBasis[128 * SB];
    __shared__ __attribute__((aligned(16))) _Float16 sAct[128 * SA];
    __shared__ float sR[128];
    __shared__ float sw2[112];
    __shared__ float sOut[3];

    const int t    = threadIdx.x;
    const int z    = blockIdx.x >> 10;
    const int a    = blockIdx.x & 1023;
    const int lane = t & 31;
    const int w    = t >> 5;       // wave id 0..7 -> rows 16w..16w+15
    const int g    = lane >> 4;    // lane half-group
    const int ln   = lane & 15;

    // ---- stage scaled W0^T (30x100 -> padded 112x32) and w2 into LDS ----
    for (int i = t; i < 112 * 32; i += 256) {
        int n = i >> 5, k = i & 31;
        float v = (n < 100 && k < 30) ? W0[k * 100 + n] * RS30 : 0.0f;
        sW0T[n * SB + k] = (_Float16)v;
    }
    // zero K-padding columns 112..127 of sAct (read by layer-1 A-fragments,
    // never written by the N=112 pipeline)
    for (int i = t; i < 128 * 16; i += 256) {
        int row = i >> 4, c = 112 + (i & 15);
        sAct[row * SA + c] = (_Float16)0.0f;
    }
    if (t < 112) sw2[t] = (t < 100) ? W2[t] * RS100 : 0.0f;
    if (t == 0) { sOut[0] = 0.0f; sOut[1] = 0.0f; sOut[2] = 0.0f; }

    const float* ga = geometry + ((size_t)z * 1024 + a) * 3;
    const float gax = ga[0], gay = ga[1], gaz = ga[2];

    float acc0 = 0.0f, acc1 = 0.0f, acc2 = 0.0f;
    float relx = 0.0f, rely = 0.0f, relz = 0.0f, featb = 0.0f;

    __syncthreads();

    const int rowA = w * 16 + ln;      // this lane's A-fragment row

    for (int chunk = 0; chunk < 8; ++chunk) {
        const int b0 = chunk * 128;

        // ---- phase A: rel, r per pair (threads 0..127 own pair t) ----
        if (t < 128) {
            const float* gp = geometry + ((size_t)z * 1024 + b0 + t) * 3;
            relx = gp[0] - gax; rely = gp[1] - gay; relz = gp[2] - gaz;
            featb = features[(size_t)z * 1024 + b0 + t];
            float r2 = relx * relx + rely * rely + relz * relz;
            sR[t] = (r2 > 1e-12f) ? __fsqrt_rn(r2) : 0.0f;
        }
        __syncthreads();

        // ---- phase B: cosine basis -> f16 LDS (all 256 threads) ----
        {
            int p  = t >> 1;
            int k0 = (t & 1) * 16;
            float dbase = sR[p] * 14.5f;        // r / step, step = 2/29
            #pragma unroll
            for (int kk = 0; kk < 16; ++kk) {
                int k = k0 + kk;
                float bval = 0.0f;
                if (k < 30) {
                    float d = dbase - (float)k;
                    if (fabsf(d) < 1.0f) {
                        float c = __cosf(1.5707963268f * d);
                        bval = c * c;
                    }
                }
                sBasis[p * SB + k] = (_Float16)bval;
            }
        }
        __syncthreads();

        // ---- layer 0: [16x32] x [32x112] per wave, 7 WMMAs ----
        AF a0;
        a0.h[0] = *(const v8h*)&sBasis[rowA * SB + g * 8];
        a0.h[1] = *(const v8h*)&sBasis[rowA * SB + 16 + g * 8];
        v8f c0[NT];
        #pragma unroll
        for (int nt = 0; nt < NT; ++nt) {
            AF b;
            const _Float16* bp = &sW0T[(nt * 16 + ln) * SB + g * 8];
            b.h[0] = *(const v8h*)bp;
            b.h[1] = *(const v8h*)(bp + 16);
            v8f zc = {};
            c0[nt] = __builtin_amdgcn_wmma_f32_16x16x32_f16(
                false, a0.v, false, b.v, (short)0, zc, false, false);
        }
        // softplus -> f16 activations (rows 16w..16w+15 are wave-private)
        #pragma unroll
        for (int nt = 0; nt < NT; ++nt) {
            #pragma unroll
            for (int j = 0; j < 8; ++j) {
                float v = actf(c0[nt][j]);
                sAct[(w * 16 + j + 8 * g) * SA + nt * 16 + ln] = (_Float16)v;
            }
        }

        // ---- layer 1: [16x128] x [128x112] per wave, 28 WMMAs ----
        v8f c1[NT];
        #pragma unroll
        for (int nt = 0; nt < NT; ++nt) { v8f zc = {}; c1[nt] = zc; }
        #pragma unroll
        for (int ks = 0; ks < 4; ++ks) {
            AF a1;
            a1.h[0] = *(const v8h*)&sAct[rowA * SA + ks * 32 + g * 8];
            a1.h[1] = *(const v8h*)&sAct[rowA * SA + ks * 32 + 16 + g * 8];
            #pragma unroll
            for (int nt = 0; nt < NT; ++nt) {
                AF b;
                const _Float16* bp = W1T + (nt * 16 + ln) * 128 + ks * 32 + g * 8;
                b.h[0] = *(const v8h*)bp;
                b.h[1] = *(const v8h*)(bp + 16);
                c1[nt] = __builtin_amdgcn_wmma_f32_16x16x32_f16(
                    false, a1.v, false, b.v, (short)0, c1[nt], false, false);
            }
        }
        // softplus -> f16 activations for layer 2 (columns 0..111 only;
        // columns 112..127 stay zero from the one-time init)
        #pragma unroll
        for (int nt = 0; nt < NT; ++nt) {
            #pragma unroll
            for (int j = 0; j < 8; ++j) {
                float v = actf(c1[nt][j]);
                sAct[(w * 16 + j + 8 * g) * SA + nt * 16 + ln] = (_Float16)v;
            }
        }
        __syncthreads();

        // ---- layer 2 + spherical-harmonic accumulation ----
        if (t < 128) {
            float R = 0.0f;
            const _Float16* ap = &sAct[t * SA];
            #pragma unroll 16
            for (int h = 0; h < 112; ++h) R += (float)ap[h] * sw2[h];
            float r = sR[t];
            if (r > 0.0f) {
                float s = R * SQRT3 * featb / r;
                acc0 += s * relx;
                acc1 += s * rely;
                acc2 += s * relz;
            }
        }
        __syncthreads();   // protect sR/sBasis/sAct for next chunk
    }

    // ---- block reduction of the 3 output components ----
    if (t < 128) {
        atomicAdd(&sOut[0], acc0);
        atomicAdd(&sOut[1], acc1);
        atomicAdd(&sOut[2], acc2);
    }
    __syncthreads();
    if (t == 0) {
        float* op = out + ((size_t)z * 1024 + a) * 3;
        op[0] = sOut[0]; op[1] = sOut[1]; op[2] = sOut[2];
    }
}

extern "C" void kernel_launch(void* const* d_in, const int* in_sizes, int n_in,
                              void* d_out, int out_size, void* d_ws, size_t ws_size,
                              hipStream_t stream) {
    const float* features = (const float*)d_in[0];   // [2,1024,1]
    const float* geometry = (const float*)d_in[1];   // [2,1024,3]
    const float* W0       = (const float*)d_in[2];   // [30,100]
    const float* W1       = (const float*)d_in[3];   // [100,100]
    const float* W2       = (const float*)d_in[4];   // [100,1]
    float*       out      = (float*)d_out;           // [2,1024,3]

    _Float16* W1T = (_Float16*)d_ws;                 // 128*128 f16 = 32 KB

    grav_prep_w1<<<64, 256, 0, stream>>>(W1, W1T);
    grav_main<<<2048, 256, 0, stream>>>(features, geometry, W0, W2, W1T, out);
}
